// SegmentVQVAE_38457137168618
// MI455X (gfx1250) — matrix operand order repaired and checked by reading, not compiled, not run on hardware
//
#include <hip/hip_runtime.h>
#include <hip/hip_bf16.h>
#include <stdint.h>

typedef __attribute__((ext_vector_type(16))) _Float16 v16h;
typedef __attribute__((ext_vector_type(8)))  _Float16 v8h;
typedef __attribute__((ext_vector_type(8)))  float    v8f;
typedef __attribute__((ext_vector_type(4)))  int      v4i;

#define Bn   32
#define Tn   256
#define Cn   8
#define Vn   1024
#define En   128
#define Dn   128
#define Kcb  512
#define BPn  4
#define NBn  24
#define FFn  512
#define Rn   8

// ---------------- CDNA5 async-to-LDS primitives ----------------
#if __has_builtin(__builtin_amdgcn_global_load_async_to_lds_b128)
#define HAVE_ASYNC_B128 1
#else
#define HAVE_ASYNC_B128 0
#endif
#if __has_builtin(__builtin_amdgcn_s_wait_asynccnt)
#define HAVE_WAIT_ASYNC 1
#else
#define HAVE_WAIT_ASYNC 0
#endif

typedef __attribute__((address_space(1))) v4i gv4i_t;
typedef __attribute__((address_space(3))) v4i sv4i_t;

__device__ __forceinline__ void async_ld16(const _Float16* g, unsigned lds_off) {
#if HAVE_ASYNC_B128
    __builtin_amdgcn_global_load_async_to_lds_b128(
        (gv4i_t*)(uintptr_t)g,
        (sv4i_t*)(uintptr_t)lds_off,
        0, 0);
#else
    asm volatile("global_load_async_to_lds_b128 %0, %1, off"
                 :: "v"(lds_off), "v"(g)
                 : "memory");
#endif
}

__device__ __forceinline__ void wait_async0() {
#if HAVE_WAIT_ASYNC
    __builtin_amdgcn_s_wait_asynccnt(0);
#else
    asm volatile("s_wait_asynccnt 0x0" ::: "memory");
#endif
}

// ---------------- init ----------------
__global__ void init_kernel(float* acc) {
    if (threadIdx.x < 2) acc[threadIdx.x] = 0.0f;
}

// ---------------- weight packing ----------------
__global__ __launch_bounds__(256) void packw1_kernel(const float* __restrict__ w1,
                                                     _Float16* __restrict__ w1p) {
    int base = (blockIdx.x * 256 + threadIdx.x) * 4;
#pragma unroll
    for (int i = 0; i < 4; ++i) {
        int idx = base + i;                 // (kw*512 + co)*1024 + ci
        int kw  = idx >> 19;
        int rem = idx & ((512 * 1024) - 1);
        int co  = rem >> 10;
        int ci  = rem & 1023;
        w1p[idx] = (_Float16)w1[(co * 1024 + ci) * 3 + kw];
    }
}

__global__ __launch_bounds__(256) void packw2_kernel(const float* __restrict__ w2,
                                                     _Float16* __restrict__ w2p) {
    int base = (blockIdx.x * 256 + threadIdx.x) * 4;
#pragma unroll
    for (int i = 0; i < 4; ++i) {
        int idx = base + i;
        w2p[idx] = (_Float16)w2[idx];
    }
}

// ---------------- embedding gather -> padded transposed activations ----------------
__global__ __launch_bounds__(256) void gather_kernel(const int* __restrict__ tokens,
                                                     const float* __restrict__ emb,
                                                     _Float16* __restrict__ xT) {
    int row = blockIdx.x;
    int b   = row / 258;
    int tp  = row % 258;
    _Float16* dst = xT + (size_t)row * 1024;
    int tid = threadIdx.x;
    if (tp == 0 || tp == 257) {
        for (int i = tid; i < 1024; i += 256) dst[i] = (_Float16)0.0f;
        return;
    }
    int t = tp - 1;
    for (int i = tid; i < 1024; i += 256) {
        int c = i >> 7;
        int e = i & 127;
        int tk = tokens[(b * Tn + t) * Cn + c];
        dst[i] = (_Float16)emb[tk * En + e];
    }
}

// ---------------- conv1 as WMMA GEMM with async-LDS B staging ----------------
// M=512 (co), N=8192 (b,t), K=3x1024. Block: 8 waves -> 64x128 tile.
// B tile (32K x 128N) staged to LDS, padded rows of 40 halves (bank stride 20).
__global__ __launch_bounds__(256) void conv1_kernel(const _Float16* __restrict__ w1p,
                                                    const _Float16* __restrict__ xT,
                                                    const float* __restrict__ b1,
                                                    _Float16* __restrict__ y1T) {
    __shared__ __align__(16) _Float16 bstage[2][128 * 40];   // 2 x 10 KB
    unsigned tid  = threadIdx.x;
    unsigned wv   = tid >> 5;
    unsigned lane = tid & 31u;
    unsigned wm   = wv & 3u;
    unsigned wn   = wv >> 2;
    int co0 = blockIdx.x * 64 + wm * 16;
    int nb0 = blockIdx.y * 128 + wn * 64;
    int ln  = lane & 15;
    int hf  = lane >> 4;

    // staging: this thread owns chunks tid and tid+256 of 512 (chunk = n*4 + part)
    int n0 = (int)(tid >> 2),          p0 = (int)(tid & 3);
    int n1 = (int)((tid + 256) >> 2),  p1 = (int)((tid + 256) & 3);
    int ng0 = blockIdx.y * 128 + n0, ng1 = blockIdx.y * 128 + n1;
    const _Float16* g0 = xT + ((size_t)((ng0 >> 8) * 258 + (ng0 & 255))) * 1024 + p0 * 8;
    const _Float16* g1 = xT + ((size_t)((ng1 >> 8) * 258 + (ng1 & 255))) * 1024 + p1 * 8;
    unsigned lbase = (unsigned)(uintptr_t)&bstage[0][0];
    unsigned l0 = lbase + (unsigned)(n0 * 40 + p0 * 8) * 2;
    unsigned l1 = lbase + (unsigned)(n1 * 40 + p1 * 8) * 2;

    v8f acc[4];
#pragma unroll
    for (int s = 0; s < 4; ++s)
#pragma unroll
        for (int r = 0; r < 8; ++r) acc[s][r] = 0.0f;

    int mrow  = co0 + ln;
    int khalf = hf * 8;
    int bidx[4], trow[4];
#pragma unroll
    for (int s = 0; s < 4; ++s) {
        int n = nb0 + s * 16 + ln;
        bidx[s] = n >> 8;
        trow[s] = n & 255;
    }

    // prologue: stage step 0 (kw=0, kb=0)
    async_ld16(g0, l0);
    async_ld16(g1, l1);

    for (int st = 0; st < 96; ++st) {
        wait_async0();          // stage(st) landed (per wave)
        __syncthreads();        // all waves: tile ready, prior tile reads done
        if (st + 1 < 96) {
            int kwn = (st + 1) >> 5;
            int kbn = ((st + 1) & 31) * 32;
            unsigned boff = (unsigned)(((st + 1) & 1) * (128 * 40 * 2));
            async_ld16(g0 + kwn * 1024 + kbn, l0 + boff);
            async_ld16(g1 + kwn * 1024 + kbn, l1 + boff);
        }
        int kw = st >> 5;
        int kb = (st & 31) * 32;
        const _Float16* Ab = w1p + ((size_t)(kw * 512 + mrow)) * 1024 + khalf + kb;
        v8h alo = *(const v8h*)(Ab);
        v8h ahi = *(const v8h*)(Ab + 16);
        v16h a;
#pragma unroll
        for (int i = 0; i < 8; ++i) { a[i] = alo[i]; a[i + 8] = ahi[i]; }
        const _Float16* sb = &bstage[st & 1][0];
#pragma unroll
        for (int s = 0; s < 4; ++s) {
            int nl = (int)wn * 64 + s * 16 + ln;
            v8h blo = *(const v8h*)(sb + nl * 40 + hf * 8);
            v8h bhi = *(const v8h*)(sb + nl * 40 + hf * 8 + 16);
            v16h bf;
#pragma unroll
            for (int i = 0; i < 8; ++i) { bf[i] = blo[i]; bf[i + 8] = bhi[i]; }
            acc[s] = __builtin_amdgcn_wmma_f32_16x16x32_f16(
                false, a, false, bf, (short)0, acc[s], false, false);
        }
    }

    int mbase = co0 + hf * 8;
#pragma unroll
    for (int s = 0; s < 4; ++s) {
        v8h o;
#pragma unroll
        for (int r = 0; r < 8; ++r) {
            float v = acc[s][r] + b1[mbase + r];
            o[r] = (_Float16)(v > 0.0f ? v : 0.0f);
        }
        *(v8h*)(y1T + ((size_t)(bidx[s] * 256 + trow[s])) * 512 + mbase) = o;
    }
}

// ---------------- conv2 (1x1) as WMMA GEMM with async-LDS B staging ----------------
// M=256 (c2), N=8192, K=512.
__global__ __launch_bounds__(256) void conv2_kernel(const _Float16* __restrict__ w2p,
                                                    const _Float16* __restrict__ y1T,
                                                    const float* __restrict__ b2,
                                                    _Float16* __restrict__ y2T) {
    __shared__ __align__(16) _Float16 bstage[2][128 * 40];
    unsigned tid  = threadIdx.x;
    unsigned wv   = tid >> 5;
    unsigned lane = tid & 31u;
    unsigned wm   = wv & 3u;
    unsigned wn   = wv >> 2;
    int co0 = blockIdx.x * 64 + wm * 16;
    int nb0 = blockIdx.y * 128 + wn * 64;
    int ln  = lane & 15;
    int hf  = lane >> 4;

    int n0 = (int)(tid >> 2),          p0 = (int)(tid & 3);
    int n1 = (int)((tid + 256) >> 2),  p1 = (int)((tid + 256) & 3);
    int ng0 = blockIdx.y * 128 + n0, ng1 = blockIdx.y * 128 + n1;
    const _Float16* g0 = y1T + ((size_t)((ng0 >> 8) * 256 + (ng0 & 255))) * 512 + p0 * 8;
    const _Float16* g1 = y1T + ((size_t)((ng1 >> 8) * 256 + (ng1 & 255))) * 512 + p1 * 8;
    unsigned lbase = (unsigned)(uintptr_t)&bstage[0][0];
    unsigned l0 = lbase + (unsigned)(n0 * 40 + p0 * 8) * 2;
    unsigned l1 = lbase + (unsigned)(n1 * 40 + p1 * 8) * 2;

    v8f acc[4];
#pragma unroll
    for (int s = 0; s < 4; ++s)
#pragma unroll
        for (int r = 0; r < 8; ++r) acc[s][r] = 0.0f;

    int mrow  = co0 + ln;
    int khalf = hf * 8;
    int bidx[4], trow[4];
#pragma unroll
    for (int s = 0; s < 4; ++s) {
        int n = nb0 + s * 16 + ln;
        bidx[s] = n >> 8;
        trow[s] = n & 255;
    }

    async_ld16(g0, l0);
    async_ld16(g1, l1);

    for (int st = 0; st < 16; ++st) {
        wait_async0();
        __syncthreads();
        if (st + 1 < 16) {
            int kbn = (st + 1) * 32;
            unsigned boff = (unsigned)(((st + 1) & 1) * (128 * 40 * 2));
            async_ld16(g0 + kbn, l0 + boff);
            async_ld16(g1 + kbn, l1 + boff);
        }
        int kb = st * 32;
        const _Float16* Ab = w2p + (size_t)mrow * 512 + khalf + kb;
        v8h alo = *(const v8h*)(Ab);
        v8h ahi = *(const v8h*)(Ab + 16);
        v16h a;
#pragma unroll
        for (int i = 0; i < 8; ++i) { a[i] = alo[i]; a[i + 8] = ahi[i]; }
        const _Float16* sb = &bstage[st & 1][0];
#pragma unroll
        for (int s = 0; s < 4; ++s) {
            int nl = (int)wn * 64 + s * 16 + ln;
            v8h blo = *(const v8h*)(sb + nl * 40 + hf * 8);
            v8h bhi = *(const v8h*)(sb + nl * 40 + hf * 8 + 16);
            v16h bf;
#pragma unroll
            for (int i = 0; i < 8; ++i) { bf[i] = blo[i]; bf[i + 8] = bhi[i]; }
            acc[s] = __builtin_amdgcn_wmma_f32_16x16x32_f16(
                false, a, false, bf, (short)0, acc[s], false, false);
        }
    }

    int mbase = co0 + hf * 8;
#pragma unroll
    for (int s = 0; s < 4; ++s) {
        v8h o;
#pragma unroll
        for (int r = 0; r < 8; ++r) {
            float v = acc[s][r] + b2[mbase + r];
            o[r] = (_Float16)(v > 0.0f ? v : 0.0f);
        }
        *(v8h*)(y2T + ((size_t)(bidx[s] * 256 + trow[s])) * 256 + mbase) = o;
    }
}

// ---------------- adaptive avg pool (T -> BP chunks of 64) ----------------
__global__ __launch_bounds__(256) void pool_kernel(const _Float16* __restrict__ y2T,
                                                   float* __restrict__ zpool_seg) {
    int idx = blockIdx.x * 256 + threadIdx.x;     // < 32*4*256
    int c2  = idx & 255;
    int rst = idx >> 8;
    int bp  = rst & 3;
    int b   = rst >> 2;
    float s = 0.0f;
    for (int i = 0; i < 64; ++i)
        s += (float)y2T[((size_t)(b * 256 + bp * 64 + i)) * 256 + c2];
    zpool_seg[idx] = s * (1.0f / 64.0f);
}

// ---------------- VQ: argmin over codebook + commit loss + pos add ----------------
__global__ __launch_bounds__(128) void vq_kernel(const float* __restrict__ zpool,
                                                 const float* __restrict__ cbi,
                                                 const float* __restrict__ cbv,
                                                 const float* __restrict__ pos,
                                                 float* __restrict__ ztf,
                                                 float* __restrict__ commit_acc) {
    __shared__ float z[128];
    __shared__ float sb[128];
    __shared__ int   si[128];
    int tid = threadIdx.x;
    int p = blockIdx.x;
    int h  = p & 1;  p >>= 1;
    int bp = p & 3;  p >>= 2;
    int b  = p & 31;
    int seg = p >> 5;

    const float* zr = zpool + (((seg * 32 + b) * 4 + bp) * 256) + h * 128;
    z[tid] = zr[tid];
    __syncthreads();

    const float* cb = h ? cbv : cbi;
    float best = 3.4e38f; int bi = 0;
#pragma unroll
    for (int i = 0; i < 4; ++i) {
        int k = tid + 128 * i;
        const float* cr = cb + (size_t)k * 128;
        float a = 0.0f;
        for (int d = 0; d < 128; ++d) {
            float cv = cr[d];
            a += cv * (cv - 2.0f * z[d]);
        }
        if (a < best || (a == best && k < bi)) { best = a; bi = k; }
    }
    sb[tid] = best; si[tid] = bi;
    __syncthreads();
    for (int s = 64; s > 0; s >>= 1) {
        if (tid < s) {
            if (sb[tid + s] < sb[tid] ||
                (sb[tid + s] == sb[tid] && si[tid + s] < si[tid])) {
                sb[tid] = sb[tid + s]; si[tid] = si[tid + s];
            }
        }
        __syncthreads();
    }
    int code = si[0];
    __syncthreads();

    float zq = cb[(size_t)code * 128 + tid];
    float diff = z[tid] - zq;
    sb[tid] = diff * diff;
    __syncthreads();
    for (int s = 64; s > 0; s >>= 1) {
        if (tid < s) sb[tid] += sb[tid + s];
        __syncthreads();
    }
    if (tid == 0) atomicAdd(commit_acc, sb[0]);

    int n = seg * 8 + h * 4 + bp;
    ztf[((size_t)b * NBn + n) * Dn + tid] = zq + pos[n * Dn + tid];
}

// ---------------- single transformer encoder layer (post-norm, relu) ----------------
__global__ __launch_bounds__(128) void tf_kernel(const float* __restrict__ ztf,
                                                 const float* __restrict__ Wqkv,
                                                 const float* __restrict__ bqkv,
                                                 const float* __restrict__ Wo,
                                                 const float* __restrict__ bo,
                                                 const float* __restrict__ g1,
                                                 const float* __restrict__ be1,
                                                 const float* __restrict__ W1,
                                                 const float* __restrict__ bb1,
                                                 const float* __restrict__ W2,
                                                 const float* __restrict__ bb2,
                                                 const float* __restrict__ g2,
                                                 const float* __restrict__ be2,
                                                 float* __restrict__ flatb) {
    __shared__ float xs[NBn * Dn];       // 12 KB
    __shared__ float buf[NBn * FFn];     // 48 KB
    __shared__ float mu[NBn], rstd[NBn];
    int b = blockIdx.x, tid = threadIdx.x;

    for (int i = tid; i < NBn * Dn; i += 128) xs[i] = ztf[(size_t)b * NBn * Dn + i];
    __syncthreads();

    for (int task = tid; task < NBn * 384; task += 128) {
        int s = task / 384, j = task % 384;
        const float* wr = Wqkv + (size_t)j * Dn;
        float a = bqkv[j];
        for (int d = 0; d < Dn; ++d) a += xs[s * Dn + d] * wr[d];
        buf[s * 384 + j] = a;
    }
    __syncthreads();

    float* obuf = buf + NBn * 384;
    if (tid < 96) {
        int s = tid >> 2, h = tid & 3;
        float sc[NBn];
        float mx = -1e30f;
        for (int u = 0; u < NBn; ++u) {
            float a = 0.0f;
            for (int j = 0; j < 32; ++j)
                a += buf[s * 384 + h * 32 + j] * buf[u * 384 + 128 + h * 32 + j];
            sc[u] = a * 0.17677669529663687f;
            mx = fmaxf(mx, sc[u]);
        }
        float se = 0.0f;
        for (int u = 0; u < NBn; ++u) { sc[u] = __expf(sc[u] - mx); se += sc[u]; }
        float inv = 1.0f / se;
        for (int j = 0; j < 32; ++j) {
            float a = 0.0f;
            for (int u = 0; u < NBn; ++u) a += sc[u] * buf[u * 384 + 256 + h * 32 + j];
            obuf[s * Dn + h * 32 + j] = a * inv;
        }
    }
    __syncthreads();

    for (int task = tid; task < NBn * Dn; task += 128) {
        int s = task / Dn, d = task % Dn;
        float a = bo[d] + xs[task];
        const float* wr = Wo + (size_t)d * Dn;
        for (int k = 0; k < Dn; ++k) a += obuf[s * Dn + k] * wr[k];
        xs[task] = a;
    }
    __syncthreads();

    if (tid < NBn) {
        float m = 0.0f;
        for (int d = 0; d < Dn; ++d) m += xs[tid * Dn + d];
        m *= (1.0f / Dn);
        float v = 0.0f;
        for (int d = 0; d < Dn; ++d) { float t = xs[tid * Dn + d] - m; v += t * t; }
        v *= (1.0f / Dn);
        mu[tid] = m; rstd[tid] = rsqrtf(v + 1e-5f);
    }
    __syncthreads();
    for (int task = tid; task < NBn * Dn; task += 128) {
        int s = task / Dn, d = task % Dn;
        xs[task] = (xs[task] - mu[s]) * rstd[s] * g1[d] + be1[d];
    }
    __syncthreads();

    for (int task = tid; task < NBn * FFn; task += 128) {
        int s = task / FFn, f = task % FFn;
        const float* wr = W1 + (size_t)f * Dn;
        float a = bb1[f];
        for (int d = 0; d < Dn; ++d) a += xs[s * Dn + d] * wr[d];
        buf[task] = a > 0.0f ? a : 0.0f;
    }
    __syncthreads();

    for (int task = tid; task < NBn * Dn; task += 128) {
        int s = task / Dn, d = task % Dn;
        const float* wr = W2 + (size_t)d * FFn;
        float a = bb2[d] + xs[task];
        for (int f = 0; f < FFn; ++f) a += buf[s * FFn + f] * wr[f];
        xs[task] = a;
    }
    __syncthreads();

    if (tid < NBn) {
        float m = 0.0f;
        for (int d = 0; d < Dn; ++d) m += xs[tid * Dn + d];
        m *= (1.0f / Dn);
        float v = 0.0f;
        for (int d = 0; d < Dn; ++d) { float t = xs[tid * Dn + d] - m; v += t * t; }
        v *= (1.0f / Dn);
        mu[tid] = m; rstd[tid] = rsqrtf(v + 1e-5f);
    }
    __syncthreads();
    for (int task = tid; task < NBn * Dn; task += 128) {
        int s = task / Dn, d = task % Dn;
        flatb[(size_t)b * (NBn * Dn) + task] =
            (xs[task] - mu[s]) * rstd[s] * g2[d] + be2[d];
    }
}

// ---------------- lora8 = flat @ lora_a^T  (B x 8) ----------------
__global__ __launch_bounds__(256) void lora_kernel(const float* __restrict__ flatb,
                                                   const float* __restrict__ lora_a,
                                                   float* __restrict__ lora8) {
    __shared__ float red[256];
    int b = blockIdx.x, tid = threadIdx.x;
    float p[8];
#pragma unroll
    for (int r = 0; r < 8; ++r) p[r] = 0.0f;
    for (int idx = tid; idx < 3072; idx += 256) {
        float f = flatb[(size_t)b * 3072 + idx];
#pragma unroll
        for (int r = 0; r < 8; ++r) p[r] += f * lora_a[r * 3072 + idx];
    }
    for (int r = 0; r < 8; ++r) {
        red[tid] = p[r];
        __syncthreads();
        for (int s = 128; s > 0; s >>= 1) {
            if (tid < s) red[tid] += red[tid + s];
            __syncthreads();
        }
        if (tid == 0) lora8[b * 8 + r] = red[0];
        __syncthreads();
    }
}

// ---------------- fused logits + log_softmax + NLL ----------------
__global__ __launch_bounds__(256) void nll_kernel(const float* __restrict__ lora_b,
                                                  const float* __restrict__ lora8,
                                                  const int* __restrict__ tok_curr,
                                                  float* __restrict__ acc) {
    __shared__ float lb[Vn * 8];       // 32 KB
    __shared__ float red[256];
    __shared__ float s_target;
    int tid = threadIdx.x;
    int rc = blockIdx.x;               // t*8 + c
    const float4* src = (const float4*)(lora_b + (size_t)rc * Vn * 8);
    float4* dst = (float4*)lb;
    for (int i = tid; i < Vn * 2; i += 256) dst[i] = src[i];
    __syncthreads();

    int t = rc >> 3, c = rc & 7;
    float nll_sum = 0.0f;
    for (int b = 0; b < Bn; ++b) {
        float l8[8];
#pragma unroll
        for (int r = 0; r < 8; ++r) l8[r] = lora8[b * 8 + r];
        float lg[4];
        float mx = -1e30f;
#pragma unroll
        for (int i = 0; i < 4; ++i) {
            int v = tid + 256 * i;
            const float* row = lb + v * 8;
            float a = 0.0f;
#pragma unroll
            for (int r = 0; r < 8; ++r) a += row[r] * l8[r];
            lg[i] = a;
            mx = fmaxf(mx, a);
        }
        red[tid] = mx;
        __syncthreads();
        for (int s = 128; s > 0; s >>= 1) {
            if (tid < s) red[tid] = fmaxf(red[tid], red[tid + s]);
            __syncthreads();
        }
        float gmx = red[0];
        __syncthreads();
        float se = 0.0f;
#pragma unroll
        for (int i = 0; i < 4; ++i) se += __expf(lg[i] - gmx);
        red[tid] = se;
        __syncthreads();
        for (int s = 128; s > 0; s >>= 1) {
            if (tid < s) red[tid] += red[tid + s];
            __syncthreads();
        }
        float gse = red[0];
        int target = tok_curr[(b * Tn + t) * Cn + c];
#pragma unroll
        for (int i = 0; i < 4; ++i) {
            int v = tid + 256 * i;
            if (v == target) s_target = lg[i];
        }
        __syncthreads();
        if (tid == 0) nll_sum += (logf(gse) + gmx - s_target);
        __syncthreads();
    }
    if (tid == 0) atomicAdd(acc, nll_sum);
}

// ---------------- finalize ----------------
__global__ void fin_kernel(const float* __restrict__ acc, float* __restrict__ out) {
    out[0] = acc[0] * (1.0f / 65536.0f) + 0.05f * acc[1] * (1.0f / 16384.0f);
}

extern "C" void kernel_launch(void* const* d_in, const int* in_sizes, int n_in,
                              void* d_out, int out_size, void* d_ws, size_t ws_size,
                              hipStream_t stream) {
    const int*   tok_prev = (const int*)d_in[0];
    const int*   tok_curr = (const int*)d_in[1];
    const int*   tok_next = (const int*)d_in[2];
    const float* emb      = (const float*)d_in[3];
    const float* conv1_w  = (const float*)d_in[4];
    const float* conv1_b  = (const float*)d_in[5];
    const float* conv2_w  = (const float*)d_in[6];
    const float* conv2_b  = (const float*)d_in[7];
    const float* cbi      = (const float*)d_in[8];
    const float* cbv      = (const float*)d_in[9];
    const float* pos      = (const float*)d_in[10];
    const float* attn_in_w  = (const float*)d_in[11];
    const float* attn_in_b  = (const float*)d_in[12];
    const float* attn_out_w = (const float*)d_in[13];
    const float* attn_out_b = (const float*)d_in[14];
    const float* ln1_g    = (const float*)d_in[15];
    const float* ln1_b    = (const float*)d_in[16];
    const float* lin1_w   = (const float*)d_in[17];
    const float* lin1_b   = (const float*)d_in[18];
    const float* lin2_w   = (const float*)d_in[19];
    const float* lin2_b   = (const float*)d_in[20];
    const float* ln2_g    = (const float*)d_in[21];
    const float* ln2_b    = (const float*)d_in[22];
    const float* lora_a   = (const float*)d_in[23];
    const float* lora_b   = (const float*)d_in[24];

    char* ws = (char*)d_ws;
    size_t off = 0;
    auto alloc = [&](size_t bytes) -> void* {
        void* p = ws + off;
        off = (off + bytes + 255) & ~(size_t)255;
        return p;
    };

    _Float16* xT    = (_Float16*)alloc((size_t)Bn * 258 * 1024 * 2);
    _Float16* y1T   = (_Float16*)alloc((size_t)Bn * Tn * 512 * 2);
    _Float16* y2T   = (_Float16*)alloc((size_t)Bn * Tn * 256 * 2);
    _Float16* w1p   = (_Float16*)alloc((size_t)3 * 512 * 1024 * 2);
    _Float16* w2p   = (_Float16*)alloc((size_t)256 * 512 * 2);
    float*    zpool = (float*)alloc((size_t)3 * Bn * BPn * 256 * 4);
    float*    ztf   = (float*)alloc((size_t)Bn * NBn * Dn * 4);
    float*    flatb = (float*)alloc((size_t)Bn * NBn * Dn * 4);
    float*    lora8 = (float*)alloc((size_t)Bn * 8 * 4);
    float*    accf  = (float*)alloc(256);

    init_kernel<<<1, 32, 0, stream>>>(accf);
    packw1_kernel<<<1536, 256, 0, stream>>>(conv1_w, w1p);
    packw2_kernel<<<128, 256, 0, stream>>>(conv2_w, w2p);

    const int* toks[3] = { tok_prev, tok_curr, tok_next };
    for (int seg = 0; seg < 3; ++seg) {
        gather_kernel<<<Bn * 258, 256, 0, stream>>>(toks[seg], emb, xT);
        conv1_kernel<<<dim3(8, 64), 256, 0, stream>>>(w1p, xT, conv1_b, y1T);
        conv2_kernel<<<dim3(4, 64), 256, 0, stream>>>(w2p, y1T, conv2_b, y2T);
        pool_kernel<<<128, 256, 0, stream>>>(y2T, zpool + (size_t)seg * Bn * BPn * 256);
    }

    vq_kernel<<<3 * Bn * BPn * 2, 128, 0, stream>>>(zpool, cbi, cbv, pos, ztf, accf + 1);
    tf_kernel<<<Bn, 128, 0, stream>>>(ztf, attn_in_w, attn_in_b, attn_out_w, attn_out_b,
                                      ln1_g, ln1_b, lin1_w, lin1_b, lin2_w, lin2_b,
                                      ln2_g, ln2_b, flatb);
    lora_kernel<<<Bn, 256, 0, stream>>>(flatb, lora_a, lora8);
    nll_kernel<<<Tn * Cn, 256, 0, stream>>>(lora_b, lora8, tok_curr, accf);
    fin_kernel<<<1, 1, 0, stream>>>(accf, (float*)d_out);
}
